// GraphModule_30820685316766
// MI455X (gfx1250) — matrix operand order repaired and checked by reading, not compiled, not run on hardware
//
#include <hip/hip_runtime.h>

// ---------------------------------------------------------------------------
// CDNA5 (gfx1250) BERT-base encoder forward.
// Dense GEMMs + attention use V_WMMA_F32_16X16X32_BF16 (wave32, fp32 acc).
// GEMM A-operand staged with GLOBAL_LOAD_ASYNC_TO_LDS_B128 when available.
// ---------------------------------------------------------------------------

typedef __attribute__((ext_vector_type(16))) __bf16 bf16x16;
typedef __attribute__((ext_vector_type(8)))  float  f32x8;

// POD 16-byte vector types (avoid HIP vector-type ctor/union issues)
struct alignas(16) U4 { unsigned int a, b, c, d; };
struct alignas(16) F4 { float x, y, z, w; };

union FragU { bf16x16 v; U4 q[2]; };

// ---- optional CDNA5 async global->LDS path (ASYNCcnt) ---------------------
#if defined(__has_builtin)
#if __has_builtin(__builtin_amdgcn_global_load_async_to_lds_b128) && \
    __has_builtin(__builtin_amdgcn_s_wait_asynccnt)
#define USE_ASYNC_LDS 1
#endif
#endif
#ifndef USE_ASYNC_LDS
#define USE_ASYNC_LDS 0
#endif

#if USE_ASYNC_LDS
// Builtin signature (from hipcc diagnostic): both pointer args are generic
// pointers to 'int __attribute__((vector_size(16)))'.
typedef int v4i_vs __attribute__((vector_size(16)));
#endif

__device__ __forceinline__ unsigned short f2bf(float f) {
  unsigned int u = __float_as_uint(f);
  u += 0x7FFFu + ((u >> 16) & 1u);           // round-to-nearest-even
  return (unsigned short)(u >> 16);
}
__device__ __forceinline__ float bf2f(unsigned short h) {
  return __uint_as_float(((unsigned int)h) << 16);
}
__device__ __forceinline__ float gelu_exact(float x) {
  return 0.5f * x * (1.0f + erff(x * 0.70710678118654752f));
}

__device__ __forceinline__ float block_sum_256(float v, float* red) {
#pragma unroll
  for (int o = 16; o; o >>= 1) v += __shfl_xor(v, o, 32);
  const int lane = threadIdx.x & 31, wv = threadIdx.x >> 5;
  __syncthreads();
  if (lane == 0) red[wv] = v;
  __syncthreads();
  float t = red[0];
#pragma unroll
  for (int i = 1; i < 8; ++i) t += red[i];
  return t;
}

// ---------------------------------------------------------------------------
// Embedding gather + LayerNorm.  One block per token, 256 threads, D=768.
// ---------------------------------------------------------------------------
__global__ __launch_bounds__(256) void embed_ln_kernel(
    const int* __restrict__ ids, const float* __restrict__ wemb,
    const float* __restrict__ pemb, const float* __restrict__ g,
    const float* __restrict__ be, float* __restrict__ outf,
    unsigned short* __restrict__ outb)
{
  __shared__ float red[8];
  const int row = blockIdx.x;           // 0..8191 (b*512+s)
  const int s   = row & 511;
  const int tid = threadIdx.x;
  const int id  = ids[row];
  const size_t base = (size_t)row * 768;

  float xv[3]; float sum = 0.0f;
#pragma unroll
  for (int j = 0; j < 3; ++j) {
    const int c = tid + j * 256;
    xv[j] = wemb[(size_t)id * 768 + c] + pemb[(size_t)s * 768 + c];
    sum += xv[j];
  }
  const float mean = block_sum_256(sum, red) * (1.0f / 768.0f);
  float v2 = 0.0f;
#pragma unroll
  for (int j = 0; j < 3; ++j) { const float d = xv[j] - mean; v2 += d * d; }
  const float var = block_sum_256(v2, red) * (1.0f / 768.0f);
  const float inv = rsqrtf(var + 1e-12f);
#pragma unroll
  for (int j = 0; j < 3; ++j) {
    const int c = tid + j * 256;
    const float o = (xv[j] - mean) * inv * g[c] + be[c];
    outf[base + c] = o;
    outb[base + c] = f2bf(o);
  }
}

// ---------------------------------------------------------------------------
// Residual add + LayerNorm.  One block per token.
// ---------------------------------------------------------------------------
__global__ __launch_bounds__(256) void residual_ln_kernel(
    const float* __restrict__ xin, const float* __restrict__ yin,
    const float* __restrict__ g, const float* __restrict__ be,
    float* __restrict__ outf, unsigned short* __restrict__ outb)
{
  __shared__ float red[8];
  const int row = blockIdx.x;
  const int tid = threadIdx.x;
  const size_t base = (size_t)row * 768;

  float xv[3]; float sum = 0.0f;
#pragma unroll
  for (int j = 0; j < 3; ++j) {
    const int c = tid + j * 256;
    xv[j] = xin[base + c] + yin[base + c];
    sum += xv[j];
  }
  const float mean = block_sum_256(sum, red) * (1.0f / 768.0f);
  float v2 = 0.0f;
#pragma unroll
  for (int j = 0; j < 3; ++j) { const float d = xv[j] - mean; v2 += d * d; }
  const float var = block_sum_256(v2, red) * (1.0f / 768.0f);
  const float inv = rsqrtf(var + 1e-12f);
#pragma unroll
  for (int j = 0; j < 3; ++j) {
    const int c = tid + j * 256;
    const float o = (xv[j] - mean) * inv * g[c] + be[c];
    outf[base + c] = o;
    outb[base + c] = f2bf(o);
  }
}

// ---------------------------------------------------------------------------
// NT GEMM:  C[M,N] = A_bf16[M,K] * W_f32[N,K]^T + bias,  EPI=1 -> exact GELU.
// Block tile 128x128, K-step 32, double-buffered LDS, 8 waves of 32x64.
// A tile: async DMA global->LDS if available.  W tile: fp32 load + convert
// (weight slab is L2-resident; converts co-issue with WMMA).
// ---------------------------------------------------------------------------
__device__ __forceinline__ void stage_storeA(
    unsigned short* AsL, int lr, int lk, const U4& ra0, const U4& ra1)
{
  *(U4*)&AsL[lr * 40 + lk]     = ra0;
  *(U4*)&AsL[lr * 40 + lk + 8] = ra1;
}

__device__ __forceinline__ void stage_storeB(
    unsigned short* BsL, int lr, int lk,
    const F4& w0, const F4& w1, const F4& w2, const F4& w3)
{
  union { U4 q; unsigned short s[8]; } p0, p1;
  p0.s[0] = f2bf(w0.x); p0.s[1] = f2bf(w0.y); p0.s[2] = f2bf(w0.z); p0.s[3] = f2bf(w0.w);
  p0.s[4] = f2bf(w1.x); p0.s[5] = f2bf(w1.y); p0.s[6] = f2bf(w1.z); p0.s[7] = f2bf(w1.w);
  p1.s[0] = f2bf(w2.x); p1.s[1] = f2bf(w2.y); p1.s[2] = f2bf(w2.z); p1.s[3] = f2bf(w2.w);
  p1.s[4] = f2bf(w3.x); p1.s[5] = f2bf(w3.y); p1.s[6] = f2bf(w3.z); p1.s[7] = f2bf(w3.w);
  *(U4*)&BsL[lr * 40 + lk]     = p0.q;
  *(U4*)&BsL[lr * 40 + lk + 8] = p1.q;
}

template <int EPI>
__global__ __launch_bounds__(256) void gemm_bf16_nt(
    const unsigned short* __restrict__ A,   // [M,K] bf16
    const float* __restrict__ W,            // [N,K] fp32
    const float* __restrict__ bias,         // [N]
    float* __restrict__ Cf,                 // [M,N] fp32 (or null)
    unsigned short* __restrict__ Cb,        // [M,N] bf16 (or null)
    int M, int N, int K)
{
  __shared__ unsigned short As[2][128 * 40];
  __shared__ unsigned short Bs[2][128 * 40];

  const int tid  = threadIdx.x;
  const int lane = tid & 31;
  const int wid  = tid >> 5;
  const int wm   = wid & 3;      // 4 waves along M
  const int wn   = wid >> 2;     // 2 waves along N
  const int m0   = blockIdx.y * 128;
  const int n0   = blockIdx.x * 128;

  const int lr = tid >> 1;            // staged row 0..127
  const int lk = (tid & 1) * 16;      // staged k offset 0/16

  const int aRow = lane & 15;
  const int aK8  = (lane >> 4) * 8;   // A frag: K 0-7/16-23 vs 8-15/24-31
  const int bCol = lane & 15;
  const int bK16 = (lane >> 4) * 8;   // B frag: K 0-7/16-23 vs 8-15/24-31

  const f32x8 zero8 = {0, 0, 0, 0, 0, 0, 0, 0};
  f32x8 acc[2][4];
#pragma unroll
  for (int i = 0; i < 2; ++i)
#pragma unroll
    for (int j = 0; j < 4; ++j) acc[i][j] = zero8;

  const int KB = K >> 5;

#if USE_ASYNC_LDS
  // Async DMA of the bf16 A tile straight into LDS (tracked by ASYNCcnt).
  {
    const unsigned short* gp = A + (size_t)(m0 + lr) * K + lk;
    unsigned short* lp = &As[0][lr * 40 + lk];
    __builtin_amdgcn_global_load_async_to_lds_b128((v4i_vs*)gp,       (v4i_vs*)lp,       0, 0);
    __builtin_amdgcn_global_load_async_to_lds_b128((v4i_vs*)(gp + 8), (v4i_vs*)(lp + 8), 0, 0);
  }
#else
  U4 ra0, ra1;
  {
    const unsigned short* ap = A + (size_t)(m0 + lr) * K + lk;
    ra0 = *(const U4*)ap; ra1 = *(const U4*)(ap + 8);
  }
#endif
  F4 rw0, rw1, rw2, rw3;
  {
    const float* wp = W + (size_t)(n0 + lr) * K + lk;
    rw0 = *(const F4*)wp;       rw1 = *(const F4*)(wp + 4);
    rw2 = *(const F4*)(wp + 8); rw3 = *(const F4*)(wp + 12);
  }
  stage_storeB(Bs[0], lr, lk, rw0, rw1, rw2, rw3);
#if USE_ASYNC_LDS
  __builtin_amdgcn_s_wait_asynccnt(0);
#else
  stage_storeA(As[0], lr, lk, ra0, ra1);
#endif
  __syncthreads();

  for (int kb = 0; kb < KB; ++kb) {
    const int cur = kb & 1;
    const bool more = (kb + 1) < KB;
    if (more) {
      const int kk = (kb + 1) << 5;
#if USE_ASYNC_LDS
      const unsigned short* gp = A + (size_t)(m0 + lr) * K + kk + lk;
      unsigned short* lp = &As[cur ^ 1][lr * 40 + lk];
      __builtin_amdgcn_global_load_async_to_lds_b128((v4i_vs*)gp,       (v4i_vs*)lp,       0, 0);
      __builtin_amdgcn_global_load_async_to_lds_b128((v4i_vs*)(gp + 8), (v4i_vs*)(lp + 8), 0, 0);
#else
      const unsigned short* ap = A + (size_t)(m0 + lr) * K + kk + lk;
      ra0 = *(const U4*)ap; ra1 = *(const U4*)(ap + 8);
#endif
      const float* wp = W + (size_t)(n0 + lr) * K + kk + lk;
      rw0 = *(const F4*)wp;       rw1 = *(const F4*)(wp + 4);
      rw2 = *(const F4*)(wp + 8); rw3 = *(const F4*)(wp + 12);
    }

    FragU a0, a1;
    {
      const unsigned short* p = &As[cur][(wm * 32 + aRow) * 40 + aK8];
      a0.q[0] = *(const U4*)p;  a0.q[1] = *(const U4*)(p + 16);
      p += 16 * 40;
      a1.q[0] = *(const U4*)p;  a1.q[1] = *(const U4*)(p + 16);
    }
#pragma unroll
    for (int ni = 0; ni < 4; ++ni) {
      FragU bf;
      const unsigned short* p = &Bs[cur][(wn * 64 + ni * 16 + bCol) * 40 + bK16];
      bf.q[0] = *(const U4*)p; bf.q[1] = *(const U4*)(p + 16);
      acc[0][ni] = __builtin_amdgcn_wmma_f32_16x16x32_bf16(
          false, a0.v, false, bf.v, (short)0, acc[0][ni], false, false);
      acc[1][ni] = __builtin_amdgcn_wmma_f32_16x16x32_bf16(
          false, a1.v, false, bf.v, (short)0, acc[1][ni], false, false);
    }

    if (more) {
      stage_storeB(Bs[cur ^ 1], lr, lk, rw0, rw1, rw2, rw3);
#if !USE_ASYNC_LDS
      stage_storeA(As[cur ^ 1], lr, lk, ra0, ra1);
#endif
    }
#if USE_ASYNC_LDS
    __builtin_amdgcn_s_wait_asynccnt(0);
#endif
    __syncthreads();
  }

  // Epilogue: bias (+GELU), fp32 and/or bf16 stores.
#pragma unroll
  for (int mi = 0; mi < 2; ++mi) {
#pragma unroll
    for (int ni = 0; ni < 4; ++ni) {
      const int colg = n0 + wn * 64 + ni * 16 + (lane & 15);
      const float bb = bias[colg];
#pragma unroll
      for (int r = 0; r < 8; ++r) {
        const int rowg = m0 + wm * 32 + mi * 16 + (lane >> 4) * 8 + r;
        float v = acc[mi][ni][r] + bb;
        if (EPI == 1) v = gelu_exact(v);
        const size_t idx = (size_t)rowg * N + colg;
        if (Cf) Cf[idx] = v;
        if (Cb) Cb[idx] = f2bf(v);
      }
    }
  }
}

// ---------------------------------------------------------------------------
// Fused attention: per (b, h, 64-query block).
// Preload: rel-bias head row + mask terms + FULL transposed V into LDS.
// Phase 1: S = QK^T*scale + relbias + mask  -> LDS fp32 [64][516]
// Phase 2: row softmax in LDS -> P bf16 [64][520]
// Phase 3: O = P @ V, barrier-free (Vt resident), ctx written bf16.
// LDS: 132096 + 66560 + 66560 + 1040 + 2048 = 268304 B  (< 320 KB/WGP).
// ---------------------------------------------------------------------------
#define SS_OFF   0
#define SP_OFF   (64 * 516 * 4)                    // 132096
#define SVT_OFF  (SP_OFF + 64 * 520 * 2)           // 198656
#define STAB_OFF (SVT_OFF + 64 * 520 * 2)          // 265216
#define SMSK_OFF (STAB_OFF + 260 * 4)              // 266256
#define ATTN_SMEM (SMSK_OFF + 512 * 4)             // 268304

__global__ __launch_bounds__(256) void attn_kernel(
    const unsigned short* __restrict__ Q,     // [B,S,D] bf16
    const unsigned short* __restrict__ Kmat,  // [B,S,D] bf16
    const unsigned short* __restrict__ V,     // [B,S,D] bf16
    const float* __restrict__ amask,          // [B,S]
    const float* __restrict__ relt,           // [257,12]
    unsigned short* __restrict__ ctx)         // [B,S,D] bf16
{
  extern __shared__ char smem[];
  float*          sS   = (float*)(smem + SS_OFF);            // [64][516]
  unsigned short* sP   = (unsigned short*)(smem + SP_OFF);   // [64][520]
  unsigned short* sVt  = (unsigned short*)(smem + SVT_OFF);  // [64 dh][520] keys
  float*          sTab = (float*)(smem + STAB_OFF);          // [257]
  float*          sMsk = (float*)(smem + SMSK_OFF);          // [512]

  const int q0  = blockIdx.x * 64;
  const int h   = blockIdx.y;
  const int b   = blockIdx.z;
  const int tid = threadIdx.x, lane = tid & 31, w = tid >> 5;
  const int aRow = lane & 15, aK8 = (lane >> 4) * 8;
  const int bCol = lane & 15, bK16 = (lane >> 4) * 8;
  const float scale = 0.125f;  // 1/sqrt(64)
  const f32x8 zero8 = {0, 0, 0, 0, 0, 0, 0, 0};

  // ---- Preload: bias table row, mask terms, transposed V ----
  for (int i = tid; i < 257; i += 256) sTab[i] = relt[i * 12 + h];
  for (int j = tid; j < 512; j += 256) sMsk[j] = (1.0f - amask[b * 512 + j]) * -10000.0f;
#pragma unroll 4
  for (int it = 0; it < 16; ++it) {
    const int unit = it * 256 + tid;        // 0..4095
    const int key  = unit & 511;
    const int dh0  = (unit >> 9) * 8;       // 0,8,...,56
    const unsigned short* vp = V + (size_t)(b * 512 + key) * 768 + h * 64 + dh0;
    union { U4 q; unsigned short s[8]; } u;
    u.q = *(const U4*)vp;
#pragma unroll
    for (int i = 0; i < 8; ++i) sVt[(dh0 + i) * 520 + key] = u.s[i];
  }
  __syncthreads();

  // ---- Phase 1: scores; wave w owns keys [w*64, w*64+64) ----
  const int key0 = w * 64;
  for (int ni = 0; ni < 4; ++ni) {
    f32x8 acc[4] = {zero8, zero8, zero8, zero8};
#pragma unroll
    for (int ks = 0; ks < 2; ++ks) {
      FragU bf;
      const unsigned short* kp =
          Kmat + (size_t)(b * 512 + key0 + ni * 16 + bCol) * 768 + h * 64 + ks * 32 + bK16;
      bf.q[0] = *(const U4*)kp; bf.q[1] = *(const U4*)(kp + 16);
#pragma unroll
      for (int mi = 0; mi < 4; ++mi) {
        FragU af;
        const unsigned short* qp =
            Q + (size_t)(b * 512 + q0 + mi * 16 + aRow) * 768 + h * 64 + ks * 32 + aK8;
        af.q[0] = *(const U4*)qp; af.q[1] = *(const U4*)(qp + 16);
        acc[mi] = __builtin_amdgcn_wmma_f32_16x16x32_bf16(
            false, af.v, false, bf.v, (short)0, acc[mi], false, false);
      }
    }
    const int kl = key0 + ni * 16 + (lane & 15);
    const float mterm = sMsk[kl];
#pragma unroll
    for (int mi = 0; mi < 4; ++mi) {
#pragma unroll
      for (int r = 0; r < 8; ++r) {
        const int ql = mi * 16 + (lane >> 4) * 8 + r;
        int d = kl - (q0 + ql);
        d = d < -128 ? -128 : (d > 128 ? 128 : d);
        sS[ql * 516 + kl] = acc[mi][r] * scale + sTab[d + 128] + mterm;
      }
    }
  }
  __syncthreads();

  // ---- Phase 2: softmax; wave w owns rows [w*8, w*8+8) ----
  for (int rr = 0; rr < 8; ++rr) {
    const int row = w * 8 + rr;
    float* Sr = sS + row * 516;
    unsigned short* Pr = sP + row * 520;
    float m = -3.0e38f;
    for (int j = lane; j < 512; j += 32) m = fmaxf(m, Sr[j]);
#pragma unroll
    for (int o = 16; o; o >>= 1) m = fmaxf(m, __shfl_xor(m, o, 32));
    float s = 0.0f;
    for (int j = lane; j < 512; j += 32) {
      const float e = __expf(Sr[j] - m);
      s += e;
      Pr[j] = f2bf(e);
    }
#pragma unroll
    for (int o = 16; o; o >>= 1) s += __shfl_xor(s, o, 32);
    const float inv = 1.0f / s;
    for (int j = lane; j < 512; j += 32) Pr[j] = f2bf(bf2f(Pr[j]) * inv);
  }
  __syncthreads();

  // ---- Phase 3: O = P @ V; barrier-free; wave w -> q-tile (w>>1),
  //      dh-tiles {(w&1)*2, (w&1)*2+1} ----
  const int mi3 = w >> 1;
  const int nb3 = (w & 1) * 2;
  f32x8 o0 = zero8, o1 = zero8;
  for (int ks = 0; ks < 16; ++ks) {
    FragU a;
    const unsigned short* pp = sP + (mi3 * 16 + aRow) * 520 + ks * 32 + aK8;
    a.q[0] = *(const U4*)pp; a.q[1] = *(const U4*)(pp + 16);
    FragU b0, b1;
    const unsigned short* v0 = sVt + (nb3 * 16 + bCol) * 520 + ks * 32 + bK16;
    b0.q[0] = *(const U4*)v0; b0.q[1] = *(const U4*)(v0 + 16);
    const unsigned short* v1 = v0 + 16 * 520;
    b1.q[0] = *(const U4*)v1; b1.q[1] = *(const U4*)(v1 + 16);
    o0 = __builtin_amdgcn_wmma_f32_16x16x32_bf16(false, a.v, false, b0.v, (short)0, o0, false, false);
    o1 = __builtin_amdgcn_wmma_f32_16x16x32_bf16(false, a.v, false, b1.v, (short)0, o1, false, false);
  }
#pragma unroll
  for (int r = 0; r < 8; ++r) {
    const int ql = mi3 * 16 + (lane >> 4) * 8 + r;
    const size_t base = (size_t)(b * 512 + q0 + ql) * 768 + h * 64;
    ctx[base + nb3 * 16 + (lane & 15)]       = f2bf(o0[r]);
    ctx[base + (nb3 + 1) * 16 + (lane & 15)] = f2bf(o1[r]);
  }
}

// ---------------------------------------------------------------------------
// Host orchestration.
// ---------------------------------------------------------------------------
extern "C" void kernel_launch(void* const* d_in, const int* in_sizes, int n_in,
                              void* d_out, int out_size, void* d_ws, size_t ws_size,
                              hipStream_t stream)
{
  (void)in_sizes; (void)n_in; (void)out_size; (void)ws_size;

  const int*   ids   = (const int*)  d_in[0];
  const float* amask = (const float*)d_in[1];
  const float* wemb  = (const float*)d_in[2];
  const float* pemb  = (const float*)d_in[3];
  const float* elnw  = (const float*)d_in[4];
  const float* elnb  = (const float*)d_in[5];
  const float* relt  = (const float*)d_in[6];
  const float* Wq    = (const float*)d_in[7];
  const float* bq    = (const float*)d_in[8];
  const float* Wk    = (const float*)d_in[9];
  const float* bk    = (const float*)d_in[10];
  const float* Wv    = (const float*)d_in[11];
  const float* bv    = (const float*)d_in[12];
  const float* Wo    = (const float*)d_in[13];
  const float* bo    = (const float*)d_in[14];
  const float* alnw  = (const float*)d_in[15];
  const float* alnb  = (const float*)d_in[16];
  const float* Wi    = (const float*)d_in[17];
  const float* bi    = (const float*)d_in[18];
  const float* Wd    = (const float*)d_in[19];
  const float* bd    = (const float*)d_in[20];
  const float* olnw  = (const float*)d_in[21];
  const float* olnb  = (const float*)d_in[22];

  const int NT = 16 * 512, D = 768, F = 3072, H = 12, L = 12;

  char* ws = (char*)d_ws;
  size_t off = 0;
  auto carve = [&](size_t bytes) -> char* {
    char* p = ws + off;
    off += (bytes + 255) & ~(size_t)255;
    return p;
  };
  float* hf   = (float*)carve((size_t)NT * D * 4);
  float* h1f  = (float*)carve((size_t)NT * D * 4);
  float* tmpf = (float*)carve((size_t)NT * D * 4);
  unsigned short* hb     = (unsigned short*)carve((size_t)NT * D * 2);
  unsigned short* h1b    = (unsigned short*)carve((size_t)NT * D * 2);
  unsigned short* qbuf   = (unsigned short*)carve((size_t)NT * D * 2);
  unsigned short* kbuf   = (unsigned short*)carve((size_t)NT * D * 2);
  unsigned short* vbuf   = (unsigned short*)carve((size_t)NT * D * 2);
  unsigned short* ctxb   = (unsigned short*)carve((size_t)NT * D * 2);
  unsigned short* interb = (unsigned short*)carve((size_t)NT * F * 2);

  embed_ln_kernel<<<NT, 256, 0, stream>>>(ids, wemb, pemb, elnw, elnb, hf, hb);

  const size_t WDD = (size_t)D * D, WFD = (size_t)F * D;
  const dim3 gDD(D / 128, NT / 128);
  const dim3 gFD(F / 128, NT / 128);

  for (int l = 0; l < L; ++l) {
    gemm_bf16_nt<0><<<gDD, 256, 0, stream>>>(hb, Wq + l * WDD, bq + l * D, nullptr, qbuf, NT, D, D);
    gemm_bf16_nt<0><<<gDD, 256, 0, stream>>>(hb, Wk + l * WDD, bk + l * D, nullptr, kbuf, NT, D, D);
    gemm_bf16_nt<0><<<gDD, 256, 0, stream>>>(hb, Wv + l * WDD, bv + l * D, nullptr, vbuf, NT, D, D);

    attn_kernel<<<dim3(8, H, 16), 256, ATTN_SMEM, stream>>>(qbuf, kbuf, vbuf, amask, relt, ctxb);

    gemm_bf16_nt<0><<<gDD, 256, 0, stream>>>(ctxb, Wo + l * WDD, bo + l * D, tmpf, nullptr, NT, D, D);
    residual_ln_kernel<<<NT, 256, 0, stream>>>(hf, tmpf, alnw + l * D, alnb + l * D, h1f, h1b);

    gemm_bf16_nt<1><<<gFD, 256, 0, stream>>>(h1b, Wi + l * WFD, bi + (size_t)l * F, nullptr, interb, NT, F, D);
    gemm_bf16_nt<0><<<gDD, 256, 0, stream>>>(interb, Wd + l * WFD, bd + l * D, tmpf, nullptr, NT, D, F);

    float* outp = (l == L - 1) ? (float*)d_out : hf;
    residual_ln_kernel<<<NT, 256, 0, stream>>>(h1f, tmpf, olnw + l * D, olnb + l * D, outp, hb);
  }
}
